// QLlamaAttention_68238440399491
// MI455X (gfx1250) — matrix operand order repaired and checked by reading, not compiled, not run on hardware
//
#include <hip/hip_runtime.h>
#include <hip/hip_bf16.h>

// ---------------- problem constants (match reference) ----------------
#define H_DIM   4096
#define NH      32
#define HD      128
#define SEQ     2048            // B*S rows
#define GROUP   32
#define NGROUPS (H_DIM / GROUP) // 128
// segments in groups: 0..79 -> 4-bit, 80..111 -> 6-bit, 112..127 -> 8-bit

typedef __attribute__((ext_vector_type(16))) _Float16 v16h;
typedef __attribute__((ext_vector_type(8)))  _Float16 v8h;
typedef __attribute__((ext_vector_type(8)))  float    v8f;
typedef __attribute__((ext_vector_type(4)))  int      v4i;

#ifndef __has_builtin
#define __has_builtin(x) 0
#endif

#if __has_builtin(__builtin_amdgcn_global_load_async_to_lds_b128)
#define HAVE_ASYNC_LDS 1
#else
#define HAVE_ASYNC_LDS 0
#endif

#define AS1 __attribute__((address_space(1)))
#define AS3 __attribute__((address_space(3)))

// 16-byte global->LDS copy on the async path (ASYNCcnt) when available.
static __device__ __forceinline__ void async_copy_b128(const _Float16* g, _Float16* l) {
#if HAVE_ASYNC_LDS
    __builtin_amdgcn_global_load_async_to_lds_b128(
        (AS1 v4i*)(uintptr_t)g,
        (AS3 v4i*)(uint32_t)(uintptr_t)l,   // low 32 bits of generic ptr = LDS offset
        0, 0);
#else
    *(v8h*)l = *(const v8h*)g;
#endif
}

static __device__ __forceinline__ void wait_async_copies() {
#if HAVE_ASYNC_LDS
#if __has_builtin(__builtin_amdgcn_s_wait_asynccnt)
    __builtin_amdgcn_s_wait_asynccnt(0);
#else
    asm volatile("s_wait_asynccnt 0x0" ::: "memory");
#endif
#endif
}

static __device__ __forceinline__ v8f v8f_zero() {
    v8f z; for (int i = 0; i < 8; ++i) z[i] = 0.0f; return z;
}

static __device__ __forceinline__ v16h make_v16(v8h lo, v8h hi) {
    v16h r;
    for (int i = 0; i < 8; ++i) { r[i] = lo[i]; r[i + 8] = hi[i]; }
    return r;
}

static __device__ __forceinline__ v8f wmma_f16(v16h a, v16h b, v8f c) {
    // v_wmma_f32_16x16x32_f16
    return __builtin_amdgcn_wmma_f32_16x16x32_f16(
        false, a, false, b, (short)0, c, false, false);
}

// ---------------------------------------------------------------------
// Fake-quant + channel reorder: out[row][j] = fq(in[row][reorder[j]])
// per-group (32) symmetric, bits by segment. One thread per (row, group).
// ---------------------------------------------------------------------
__global__ void quant_reorder_kernel(const float* __restrict__ in,
                                     const int* __restrict__ reorder,
                                     _Float16* __restrict__ out,
                                     int rows) {
    int idx = blockIdx.x * blockDim.x + threadIdx.x;
    int total = rows * NGROUPS;
    if (idx >= total) return;
    int row = idx / NGROUPS;
    int g   = idx - row * NGROUPS;
    float qmax = (g < 80) ? 7.0f : (g < 112) ? 31.0f : 127.0f;

    const float* irow = in + (size_t)row * H_DIM;
    int base = g * GROUP;
    float vals[GROUP];
    float amax = 0.0f;
    for (int i = 0; i < GROUP; ++i) {
        float v = irow[reorder[base + i]];
        vals[i] = v;
        amax = fmaxf(amax, fabsf(v));
    }
    float s = amax / qmax + 1e-8f;
    float inv = 1.0f / s;
    _Float16* orow = out + (size_t)row * H_DIM + base;
    for (int i = 0; i < GROUP; ++i) {
        float q = rintf(vals[i] * inv);
        q = fminf(fmaxf(q, -qmax), qmax);
        orow[i] = (_Float16)(q * s);
    }
}

// ---------------------------------------------------------------------
// C[M,N] = A[M,K] * B[N,K]^T  (f16 in, f32 out).
// Block = 256 threads = 8 waves (4x2); block tile 128x64; wave tile 32x32.
// K staged in 32-wide slabs, double-buffered in LDS via async loads.
// ---------------------------------------------------------------------
#define BM 128
#define BN 64
#define BK 32

__global__ __launch_bounds__(256)
void gemm_f16_kernel(const _Float16* __restrict__ A,
                     const _Float16* __restrict__ Bm,
                     float* __restrict__ C,
                     int M, int N, int K) {
    __shared__ __align__(16) _Float16 Ash[2][BM * BK];  // 2 x 8 KB
    __shared__ __align__(16) _Float16 Bsh[2][BN * BK];  // 2 x 4 KB

    int t    = threadIdx.x;
    int lane = t & 31;
    int w    = t >> 5;
    int half = lane >> 4;
    int l16  = lane & 15;
    int wr   = w >> 1;          // 0..3
    int wc   = w & 1;           // 0..1

    int mb = blockIdx.y * BM;
    int nb = blockIdx.x * BN;

    // staging assignment: A slab 128x32 halfs -> 2 x b128 per thread,
    //                     B slab  64x32 halfs -> 1 x b128 per thread
    int arow  = t >> 1;             // 0..127
    int apart = (t & 1) * 16;       // half offset within row
    int brow  = t >> 2;             // 0..63
    int bpart = (t & 3) * 8;
    const _Float16* agp = A  + (size_t)(mb + arow) * K + apart;
    const _Float16* bgp = Bm + (size_t)(nb + brow) * K + bpart;

    auto load_tile = [&](int buf, int kk) {
        __builtin_prefetch(agp + kk + 2 * BK, 0, 1);    // global_prefetch_b8
        _Float16* al = &Ash[buf][arow * BK + apart];
        async_copy_b128(agp + kk,     al);
        async_copy_b128(agp + kk + 8, al + 8);
        _Float16* bl = &Bsh[buf][brow * BK + bpart];
        async_copy_b128(bgp + kk, bl);
    };

    v8f acc[2][2];
    for (int i = 0; i < 2; ++i)
        for (int j = 0; j < 2; ++j) acc[i][j] = v8f_zero();

    load_tile(0, 0);
    wait_async_copies();
    __syncthreads();

    int buf = 0;
    int ka  = half * 8;     // A-layout K base within slab
    int kb  = half * 8;     // B-layout K base within slab
    int m0l = wr * 32;
    int n0l = wc * 32;

    for (int kk = 0; kk < K; kk += BK) {
        if (kk + BK < K) load_tile(buf ^ 1, kk + BK);   // prefetch next slab

        const _Float16* As = Ash[buf];
        const _Float16* Bs = Bsh[buf];
        // A layout (16-bit 16x32): lanes 0-15 K{0..7,16..23}; lanes 16-31 K{8..15,24..31}
        const _Float16* a0p = As + (m0l + l16) * BK + ka;
        const _Float16* a1p = As + (m0l + 16 + l16) * BK + ka;
        v16h a0 = make_v16(*(const v8h*)a0p, *(const v8h*)(a0p + 16));
        v16h a1 = make_v16(*(const v8h*)a1p, *(const v8h*)(a1p + 16));
        // B layout (16-bit 32x16): lanes 0-15 K{0..7,16..23}; lanes 16-31 K{8..15,24..31}
        const _Float16* b0p = Bs + (n0l + l16) * BK + kb;
        const _Float16* b1p = Bs + (n0l + 16 + l16) * BK + kb;
        v16h b0 = make_v16(*(const v8h*)b0p, *(const v8h*)(b0p + 16));
        v16h b1 = make_v16(*(const v8h*)b1p, *(const v8h*)(b1p + 16));

        acc[0][0] = wmma_f16(a0, b0, acc[0][0]);
        acc[0][1] = wmma_f16(a0, b1, acc[0][1]);
        acc[1][0] = wmma_f16(a1, b0, acc[1][0]);
        acc[1][1] = wmma_f16(a1, b1, acc[1][1]);

        wait_async_copies();
        __syncthreads();
        buf ^= 1;
    }

    // C/D 16x16 f32 layout: VGPR v -> row v + 8*half, col = l16
    for (int i = 0; i < 2; ++i) {
        for (int j = 0; j < 2; ++j) {
            int rbase = mb + m0l + i * 16 + half * 8;
            int cidx  = nb + n0l + j * 16 + l16;
            for (int v = 0; v < 8; ++v)
                C[(size_t)(rbase + v) * N + cidx] = acc[i][j][v];
        }
    }
}

// ---------------------------------------------------------------------
// RoPE (fp32) + convert q,k,v to f16 laid out [head][seq][hd].
// ---------------------------------------------------------------------
__global__ void rope_cvt_kernel(const float* __restrict__ qf,
                                const float* __restrict__ kf,
                                const float* __restrict__ vf,
                                _Float16* __restrict__ qh,
                                _Float16* __restrict__ kh,
                                _Float16* __restrict__ vh) {
    int idx = blockIdx.x * blockDim.x + threadIdx.x;
    if (idx >= SEQ * NH * (HD / 2)) return;
    int d = idx & 63;
    int h = (idx >> 6) & (NH - 1);
    int s = idx >> 11;

    // inv_freq = theta^(-2d/HD); ln(10000) = 9.210340371976184
    float inv_freq = __expf(-(2.0f * (float)d / (float)HD) * 9.210340371976184f);
    float ang = (float)s * inv_freq;
    float sn, cs;
    __sincosf(ang, &sn, &cs);

    size_t src = (size_t)s * H_DIM + (size_t)h * HD;
    size_t dst = ((size_t)h * SEQ + s) * HD;

    float q1 = qf[src + d], q2 = qf[src + d + 64];
    qh[dst + d]      = (_Float16)(q1 * cs - q2 * sn);
    qh[dst + d + 64] = (_Float16)(q2 * cs + q1 * sn);

    float k1 = kf[src + d], k2 = kf[src + d + 64];
    kh[dst + d]      = (_Float16)(k1 * cs - k2 * sn);
    kh[dst + d + 64] = (_Float16)(k2 * cs + k1 * sn);

    vh[dst + d]      = (_Float16)vf[src + d];
    vh[dst + d + 64] = (_Float16)vf[src + d + 64];
}

// ---------------------------------------------------------------------
// Flash-style causal attention. One wave (block=32) per (head, 16-query
// block). QK^T and PV via WMMA f16->f32; online softmax in f32.
// V tiles staged to LDS on the async path, overlapping QK^T + softmax.
// ---------------------------------------------------------------------
__global__ __launch_bounds__(32)
void attn_kernel(const _Float16* __restrict__ qh,
                 const _Float16* __restrict__ kh,
                 const _Float16* __restrict__ vh,
                 float* __restrict__ out) {
    __shared__ __align__(32) _Float16 vbuf[32 * HD];   // 32 keys x 128 dims
    __shared__ __align__(32) _Float16 pbuf[16 * 32];   // P tile, row-major

    int lane = threadIdx.x;
    int half = lane >> 4;
    int l16  = lane & 15;
    int qb   = blockIdx.x & 127;       // query block (16 rows)
    int h    = blockIdx.x >> 7;        // head

    const _Float16* Q  = qh + (size_t)h * SEQ * HD;
    const _Float16* Kp = kh + (size_t)h * SEQ * HD;
    const _Float16* Vp = vh + (size_t)h * SEQ * HD;

    // Q fragments: 4 chunks of 32 along HD, A-matrix layout
    v16h aQ[4];
    {
        int qrow = qb * 16 + l16;
        for (int c = 0; c < 4; ++c) {
            const _Float16* p = Q + (size_t)qrow * HD + c * 32 + half * 8;
            aQ[c] = make_v16(*(const v8h*)p, *(const v8h*)(p + 16));
        }
    }

    v8f o[8];
    for (int t = 0; t < 8; ++t) o[t] = v8f_zero();
    float mrun[8], lrun[8];
    for (int v = 0; v < 8; ++v) { mrun[v] = -1e30f; lrun[v] = 0.0f; }

    const float scale = 0.08838834764831845f;   // 1/sqrt(128)
    int ktiles = (qb >> 1) + 1;                 // 32-key tiles, causal bound

    for (int kt = 0; kt < ktiles; ++kt) {
        int k0 = kt * 32;

        // stage V tile into LDS (async): lane copies one key row (256B)
        {
            const _Float16* vr = Vp + (size_t)(k0 + lane) * HD;
            _Float16* dst = vbuf + lane * HD;
            for (int i = 0; i < 16; ++i)
                async_copy_b128(vr + i * 8, dst + i * 8);
        }

        // scores: two 16-key subtiles, 4 chained WMMAs over HD=128
        v8f st[2];
        for (int sub = 0; sub < 2; ++sub) {
            v8f c0 = v8f_zero();
            int kcol = k0 + sub * 16 + l16;     // key index (B-matrix N)
            const _Float16* kr = Kp + (size_t)kcol * HD + half * 8;
            for (int c = 0; c < 4; ++c) {
                v16h b = make_v16(*(const v8h*)(kr + c * 32), *(const v8h*)(kr + c * 32 + 16));
                c0 = wmma_f16(aQ[c], b, c0);
            }
            st[sub] = c0;
        }

        bool last = (kt == ktiles - 1);
        float alpha[8];
        for (int v = 0; v < 8; ++v) {
            int row = qb * 16 + v + half * 8;
            float s0 = st[0][v] * scale;
            float s1 = st[1][v] * scale;
            if (last) {
                if (k0 + l16 > row)      s0 = -1e30f;
                if (k0 + 16 + l16 > row) s1 = -1e30f;
            }
            float t = fmaxf(s0, s1);
            for (int m = 8; m >= 1; m >>= 1) t = fmaxf(t, __shfl_xor(t, m, 32));
            float mnew = fmaxf(mrun[v], t);
            float a  = __expf(mrun[v] - mnew);
            float p0 = __expf(s0 - mnew);
            float p1 = __expf(s1 - mnew);
            float rs = p0 + p1;
            for (int m = 8; m >= 1; m >>= 1) rs += __shfl_xor(rs, m, 32);
            lrun[v] = lrun[v] * a + rs;
            mrun[v] = mnew;
            alpha[v] = a;
            pbuf[(v + half * 8) * 32 + l16]      = (_Float16)p0;
            pbuf[(v + half * 8) * 32 + 16 + l16] = (_Float16)p1;
        }
        wait_async_copies();    // V tile landed (overlapped with the above)
        __syncthreads();

        // reload P as A-matrix fragment (16x32 f16)
        v16h aP;
        {
            const _Float16* pr = pbuf + l16 * 32 + half * 8;
            for (int i = 0; i < 8; ++i) { aP[i] = pr[i]; aP[i + 8] = pr[i + 16]; }
        }

        // PV: 8 dim tiles of 16; rescale then accumulate
        for (int t = 0; t < 8; ++t) {
            v8f acc = o[t];
            for (int v = 0; v < 8; ++v) acc[v] *= alpha[v];
            v16h bV;
            int dim = t * 16 + l16;
            for (int j = 0; j < 16; ++j)
                bV[j] = vbuf[(half * 8 + (j & 7) + (j >> 3) * 16) * HD + dim];
            o[t] = wmma_f16(aP, bV, acc);
        }
        __syncthreads();
    }

    // normalize and store (out[s][h*HD + d], fp32)
    for (int t = 0; t < 8; ++t) {
        for (int v = 0; v < 8; ++v) {
            int row = qb * 16 + v + half * 8;
            int col = h * HD + t * 16 + l16;
            out[(size_t)row * H_DIM + col] = o[t][v] / lrun[v];
        }
    }
}

// ---------------------------------------------------------------------
extern "C" void kernel_launch(void* const* d_in, const int* in_sizes, int n_in,
                              void* d_out, int out_size, void* d_ws, size_t ws_size,
                              hipStream_t stream) {
    const float* x  = (const float*)d_in[0];
    const float* Wq = (const float*)d_in[1];
    const float* Wk = (const float*)d_in[2];
    const float* Wv = (const float*)d_in[3];
    const float* Wo = (const float*)d_in[4];
    const int*   rq = (const int*)d_in[5];
    const int*   rk = (const int*)d_in[6];
    const int*   rv = (const int*)d_in[7];
    const int*   ro = (const int*)d_in[8];
    float* y = (float*)d_out;

    char* ws = (char*)d_ws;
    _Float16* wbuf = (_Float16*)(ws);                          // 32 MB
    _Float16* xbuf = (_Float16*)(ws + (size_t)33554432);       // 16 MB
    float*    qf   = (float*)   (ws + (size_t)50331648);       // 32 MB
    float*    kf   = (float*)   (ws + (size_t)83886080);       // 32 MB
    float*    vf   = (float*)   (ws + (size_t)117440512);      // 32 MB
    _Float16* qh   = (_Float16*)(ws + (size_t)150994944);      // 16 MB
    _Float16* kh   = (_Float16*)(ws + (size_t)167772160);      // 16 MB
    _Float16* vh   = (_Float16*)(ws + (size_t)184549376);      // 16 MB

    dim3 gemm_grid(H_DIM / BN, SEQ / BM);   // (64, 16)
    dim3 gemm_block(256);

    // Q projection
    quant_reorder_kernel<<<(H_DIM * NGROUPS + 255) / 256, 256, 0, stream>>>(Wq, rq, wbuf, H_DIM);
    quant_reorder_kernel<<<(SEQ   * NGROUPS + 255) / 256, 256, 0, stream>>>(x,  rq, xbuf, SEQ);
    gemm_f16_kernel<<<gemm_grid, gemm_block, 0, stream>>>(xbuf, wbuf, qf, SEQ, H_DIM, H_DIM);

    // K projection
    quant_reorder_kernel<<<(H_DIM * NGROUPS + 255) / 256, 256, 0, stream>>>(Wk, rk, wbuf, H_DIM);
    quant_reorder_kernel<<<(SEQ   * NGROUPS + 255) / 256, 256, 0, stream>>>(x,  rk, xbuf, SEQ);
    gemm_f16_kernel<<<gemm_grid, gemm_block, 0, stream>>>(xbuf, wbuf, kf, SEQ, H_DIM, H_DIM);

    // V projection
    quant_reorder_kernel<<<(H_DIM * NGROUPS + 255) / 256, 256, 0, stream>>>(Wv, rv, wbuf, H_DIM);
    quant_reorder_kernel<<<(SEQ   * NGROUPS + 255) / 256, 256, 0, stream>>>(x,  rv, xbuf, SEQ);
    gemm_f16_kernel<<<gemm_grid, gemm_block, 0, stream>>>(xbuf, wbuf, vf, SEQ, H_DIM, H_DIM);

    // RoPE + f16 conversion to [head][seq][hd]
    rope_cvt_kernel<<<(SEQ * NH * (HD / 2) + 255) / 256, 256, 0, stream>>>(qf, kf, vf, qh, kh, vh);

    // causal attention -> qf reused as attention output [seq][H]
    attn_kernel<<<NH * (SEQ / 16), 32, 0, stream>>>(qh, kh, vh, qf);

    // output projection
    quant_reorder_kernel<<<(SEQ   * NGROUPS + 255) / 256, 256, 0, stream>>>(qf, ro, xbuf, SEQ);
    quant_reorder_kernel<<<(H_DIM * NGROUPS + 255) / 256, 256, 0, stream>>>(Wo, ro, wbuf, H_DIM);
    gemm_f16_kernel<<<gemm_grid, gemm_block, 0, stream>>>(xbuf, wbuf, y, SEQ, H_DIM, H_DIM);
}
